// SelfAttention_75222057222427
// MI455X (gfx1250) — matrix operand order repaired and checked by reading, not compiled, not run on hardware
//
#include <hip/hip_runtime.h>
#include <hip/hip_bf16.h>

// ---------------------------------------------------------------------------
// Self-attention (B=4, S=2048, D=1024, single head, non-causal) for gfx1250.
// GEMMs: v_wmma_f32_16x16x32_bf16, f32 accumulate. 128-col B tiles staged to
// LDS with GLOBAL_LOAD_ASYNC_TO_LDS_B128 (ASYNCcnt double buffering, 2 chunks
// per thread per tile); all 8 waves of a block share each tile. Per-wave
// output tile is 16x128 (8 accumulators) -> 8 wmma per barrier pair.
// ---------------------------------------------------------------------------

typedef __attribute__((ext_vector_type(16))) __bf16 v16bf;
typedef __attribute__((ext_vector_type(8)))  __bf16 bf16x8;
typedef __attribute__((ext_vector_type(8)))  float  v8f;

#define BB 4
#define SS 2048
#define DD 1024

// LDS B tile: 128 cols x 32 K x 2B, stored with 16B pad per col (80B stride)
// for conflict-free ds_load_b128. Double buffered.
#define BTILE_COLS   128
#define BTILE_STRIDE 80
#define BTILE_BYTES  (BTILE_COLS * BTILE_STRIDE)  // 10240
#define SMEM_BYTES   (2 * BTILE_BYTES)            // 20480

constexpr float ATTN_SCALE = 0.03125f;            // 1/sqrt(1024)
constexpr float LOG_THETA  = 9.210340371976184f;  // ln(10000)

static __device__ __forceinline__ __bf16 f32_bf16(float f) {
  unsigned u = __builtin_bit_cast(unsigned, f);
  u += 0x7fffu + ((u >> 16) & 1u);                // round-to-nearest-even
  unsigned short h = (unsigned short)(u >> 16);
  return __builtin_bit_cast(__bf16, h);
}

// A-operand gather per ISA 16-bit A layout: lane holds row (lane&15);
// element e -> K = (e<8 ? e : e+8) + (lane>=16 ? 8 : 0).
static __device__ __forceinline__ v16bf load_a16(const __bf16* p) {
  bf16x8 lo = *(const bf16x8*)p;
  bf16x8 hi = *(const bf16x8*)(p + 16);
  return __builtin_shufflevector(lo, hi, 0, 1, 2, 3, 4, 5, 6, 7,
                                         8, 9, 10, 11, 12, 13, 14, 15);
}

static __device__ __forceinline__ v16bf load_b16_lds(const unsigned char* p) {
  bf16x8 lo = *(const bf16x8*)p;
  bf16x8 hi = *(const bf16x8*)(p + 32);
  return __builtin_shufflevector(lo, hi, 0, 1, 2, 3, 4, 5, 6, 7,
                                         8, 9, 10, 11, 12, 13, 14, 15);
}

// ---------------------------------------------------------------------------
// Block-cooperative GEMM inner loop. Each wave accumulates a 16x128 tile.
//   abase : per-lane A pointer (row resolved, +hf*8), advances 32/iter
//   bblock: base of the 128-column B block (uniform across block)
//   bstride_e: element stride between B columns (rows of the source matrix)
// B tiles (128 cols x 32 K) are pushed to LDS by async-to-LDS loads: each of
// the 256 threads moves two 16B chunks per tile.
// ---------------------------------------------------------------------------
template <int KLEN>
static __device__ __forceinline__ void gemm_lds(const __bf16* abase,
                                                const __bf16* bblock,
                                                unsigned bstride_e, v8f acc[8],
                                                unsigned char* smem, int tid) {
  const int lane = tid & 31;
  const int hf = lane >> 4;
  const int l15 = lane & 15;
  // chunk c in [0,512): col = c>>2 (0..127), kc = c&3 (16B chunk of the col's
  // 64B K-window). Thread handles chunks tid and tid+256.
  const int c0 = tid, c1 = tid + 256;
  const unsigned voff0 = (unsigned)((c0 >> 2) * (bstride_e * 2u) + (c0 & 3) * 16u);
  const unsigned voff1 = (unsigned)((c1 >> 2) * (bstride_e * 2u) + (c1 & 3) * 16u);
  const unsigned lds_base = (unsigned)(size_t)smem;
  const unsigned lds0 = lds_base + (unsigned)((c0 >> 2) * BTILE_STRIDE + (c0 & 3) * 16);
  const unsigned lds1 = lds_base + (unsigned)((c1 >> 2) * BTILE_STRIDE + (c1 & 3) * 16);
  const unsigned long long sbase = (unsigned long long)(size_t)bblock;

  // Prologue: tile 0 -> buffer 0
  asm volatile("global_load_async_to_lds_b128 %0, %1, %2" ::"v"(lds0),
               "v"(voff0), "s"(sbase) : "memory");
  asm volatile("global_load_async_to_lds_b128 %0, %1, %2" ::"v"(lds1),
               "v"(voff1), "s"(sbase) : "memory");

  constexpr int NT = KLEN / 32;
  for (int i = 0; i < NT; ++i) {
    if (i + 1 < NT) {
      const unsigned bofs = ((i + 1) & 1) ? (unsigned)BTILE_BYTES : 0u;
      const unsigned long long sb2 = sbase + (unsigned long long)(i + 1) * 64ull;
      asm volatile("global_load_async_to_lds_b128 %0, %1, %2" ::"v"(lds0 + bofs),
                   "v"(voff0), "s"(sb2) : "memory");
      asm volatile("global_load_async_to_lds_b128 %0, %1, %2" ::"v"(lds1 + bofs),
                   "v"(voff1), "s"(sb2) : "memory");
      asm volatile("s_wait_asynccnt 0x2" ::: "memory");  // own tile i landed
    } else {
      asm volatile("s_wait_asynccnt 0x0" ::: "memory");
    }
    if (i + 8 < NT)  // warm L2/WGP$ for the strided per-lane A chunks
      __builtin_prefetch(abase + (i + 8) * 32, 0, 0);
    __syncthreads();  // all waves' chunks of tile i landed

    const unsigned char* bb = smem + ((i & 1) ? BTILE_BYTES : 0);
    const v16bf A = load_a16(abase + i * 32);
#pragma unroll
    for (int j = 0; j < 8; ++j) {
      const v16bf Bm =
          load_b16_lds(bb + (j * 16 + l15) * BTILE_STRIDE + hf * 16);
      acc[j] = __builtin_amdgcn_wmma_f32_16x16x32_bf16(
          false, A, false, Bm, (short)0, acc[j], false, false);
    }
    __syncthreads();  // tile i fully consumed before buffer reuse
  }
}

// ------------------------------ converts -----------------------------------
__global__ void k_cvt_bf16(const float* __restrict__ in,
                           __bf16* __restrict__ out, int n) {
  int i = blockIdx.x * blockDim.x + threadIdx.x;
  int stride = gridDim.x * blockDim.x;
  for (; i < n; i += stride) out[i] = f32_bf16(in[i]);
}

// ------------------------- QKV projection + RoPE ---------------------------
__global__ __launch_bounds__(256) void k_qkv(const __bf16* __restrict__ xb,
                                             const __bf16* __restrict__ wb,
                                             const float* __restrict__ bias,
                                             __bf16* __restrict__ out,
                                             int mode) {
  __shared__ __align__(16) unsigned char smem[SMEM_BYTES];
  const int tid = threadIdx.x;
  const int lane = tid & 31;
  const int wv = tid >> 5;
  const int hf = lane >> 4;
  const int l15 = lane & 15;
  const int row0 = blockIdx.x * 128 + wv * 16;  // over B*S
  const int col0 = blockIdx.y * 128;            // over D

  v8f acc[8];
#pragma unroll
  for (int j = 0; j < 8; ++j) acc[j] = (v8f)0.0f;

  gemm_lds<DD>(xb + (size_t)(row0 + l15) * DD + hf * 8,
               wb + (size_t)col0 * DD, DD, acc, smem, tid);

  if (mode == 2) {  // V -> transposed store [b][d][s]
    const int bidx = row0 / SS;
    const int srow = (row0 & (SS - 1)) + hf * 8;
#pragma unroll
    for (int j = 0; j < 8; ++j) {
      const int colg = col0 + j * 16 + l15;
      const float bc = bias[colg];
      unsigned short h[8];
#pragma unroll
      for (int v = 0; v < 8; ++v)
        h[v] = __builtin_bit_cast(unsigned short, f32_bf16(acc[j][v] + bc));
      uint4 pk;
      pk.x = (unsigned)h[0] | ((unsigned)h[1] << 16);
      pk.y = (unsigned)h[2] | ((unsigned)h[3] << 16);
      pk.z = (unsigned)h[4] | ((unsigned)h[5] << 16);
      pk.w = (unsigned)h[6] | ((unsigned)h[7] << 16);
      *(uint4*)(out + ((size_t)bidx * DD + colg) * SS + srow) = pk;
    }
  } else {  // Q or K -> bias + RoPE, row-major bf16
#pragma unroll
    for (int j = 0; j < 8; ++j) {
      const int colg = col0 + j * 16 + l15;
      const float bc = bias[colg];
      const float invf = __expf((float)(colg & ~1) * (-LOG_THETA / (float)DD));
#pragma unroll
      for (int v = 0; v < 8; ++v) {
        const int row = row0 + v + hf * 8;
        float val = acc[j][v] + bc;
        float partner = __shfl_xor(val, 1, 32);  // adjacent column, wave32
        float ang = (float)(row & (SS - 1)) * invf;
        float sn, cs;
        __sincosf(ang, &sn, &cs);
        float o = (colg & 1) ? __fmaf_rn(partner, sn, val * cs)
                             : __fmaf_rn(val, cs, -(partner * sn));
        out[(size_t)row * DD + colg] = f32_bf16(o);
      }
    }
  }
}

// ------------------------------ scores = Q K^T -----------------------------
__global__ __launch_bounds__(256) void k_scores(const __bf16* __restrict__ qb,
                                                const __bf16* __restrict__ kb,
                                                float* __restrict__ sb) {
  __shared__ __align__(16) unsigned char smem[SMEM_BYTES];
  const int tid = threadIdx.x;
  const int lane = tid & 31;
  const int wv = tid >> 5;
  const int hf = lane >> 4;
  const int l15 = lane & 15;
  const int b = blockIdx.z;
  const int q0 = blockIdx.x * 128 + wv * 16;
  const int k0 = blockIdx.y * 128;

  v8f acc[8];
#pragma unroll
  for (int j = 0; j < 8; ++j) acc[j] = (v8f)0.0f;

  gemm_lds<DD>(qb + (size_t)(b * SS + q0 + l15) * DD + hf * 8,
               kb + (size_t)(b * SS + k0) * DD, DD, acc, smem, tid);

#pragma unroll
  for (int j = 0; j < 8; ++j) {
    const int kg = k0 + j * 16 + l15;
#pragma unroll
    for (int v = 0; v < 8; ++v) {
      const int qg = q0 + v + hf * 8;
      sb[((size_t)b * SS + qg) * SS + kg] = acc[j][v] * ATTN_SCALE;
    }
  }
}

// ---------------- row softmax; write P (bf16) in place over scores ---------
__global__ __launch_bounds__(256) void k_softmax(float* __restrict__ sb) {
  __shared__ float red[256];
  const int t = threadIdx.x;
  float* r = sb + (size_t)blockIdx.x * SS;

  float vals[8];
  float m = -3.0e38f;
#pragma unroll
  for (int i = 0; i < 8; ++i) {
    vals[i] = r[t + i * 256];
    m = fmaxf(m, vals[i]);
  }
  red[t] = m;
  __syncthreads();
  for (int o = 128; o > 0; o >>= 1) {
    if (t < o) red[t] = fmaxf(red[t], red[t + o]);
    __syncthreads();
  }
  m = red[0];
  __syncthreads();

  float s = 0.f;
#pragma unroll
  for (int i = 0; i < 8; ++i) {
    vals[i] = __expf(vals[i] - m);
    s += vals[i];
  }
  red[t] = s;
  __syncthreads();
  for (int o = 128; o > 0; o >>= 1) {
    if (t < o) red[t] += red[t + o];
    __syncthreads();
  }
  const float inv = 1.0f / red[0];

  // Safe aliasing: every thread's loads completed before the first barrier.
  __bf16* p = (__bf16*)r;
#pragma unroll
  for (int i = 0; i < 8; ++i) p[t + i * 256] = f32_bf16(vals[i] * inv);
}

// ------------------------------ out = P V ----------------------------------
// P rows are bf16[SS] at the start of each f32 score row (row stride 2*SS in
// bf16 elements). V is [b][d][s] so B columns are contiguous along keys.
__global__ __launch_bounds__(256) void k_pv(const float* __restrict__ sb,
                                            const __bf16* __restrict__ vt,
                                            float* __restrict__ out) {
  __shared__ __align__(16) unsigned char smem[SMEM_BYTES];
  const int tid = threadIdx.x;
  const int lane = tid & 31;
  const int wv = tid >> 5;
  const int hf = lane >> 4;
  const int l15 = lane & 15;
  const int b = blockIdx.z;
  const int q0 = blockIdx.x * 128 + wv * 16;
  const int d0 = blockIdx.y * 128;

  v8f acc[8];
#pragma unroll
  for (int j = 0; j < 8; ++j) acc[j] = (v8f)0.0f;

  const __bf16* pbase = (const __bf16*)sb;
  gemm_lds<SS>(pbase + (size_t)(b * SS + q0 + l15) * (2 * SS) + hf * 8,
               vt + (size_t)(b * DD + d0) * SS, SS, acc, smem, tid);

#pragma unroll
  for (int j = 0; j < 8; ++j) {
    const int dg = d0 + j * 16 + l15;
#pragma unroll
    for (int v = 0; v < 8; ++v) {
      const int qg = q0 + v + hf * 8;
      out[((size_t)b * SS + qg) * DD + dg] = acc[j][v];
    }
  }
}

// ---------------------------------------------------------------------------
extern "C" void kernel_launch(void* const* d_in, const int* in_sizes, int n_in,
                              void* d_out, int out_size, void* d_ws,
                              size_t ws_size, hipStream_t stream) {
  const float* x = (const float*)d_in[0];
  const float* wq = (const float*)d_in[1];
  const float* bq = (const float*)d_in[2];
  const float* wk = (const float*)d_in[3];
  const float* bk = (const float*)d_in[4];
  const float* wv = (const float*)d_in[5];
  const float* bv = (const float*)d_in[6];
  float* out = (float*)d_out;

  // Workspace layout (bytes): ~140.5 MB total
  char* ws = (char*)d_ws;
  size_t off = 0;
  __bf16* xb = (__bf16*)(ws + off);  off += (size_t)BB * SS * DD * 2;  // 16 MB
  __bf16* wqb = (__bf16*)(ws + off); off += (size_t)DD * DD * 2;       //  2 MB
  __bf16* wkb = (__bf16*)(ws + off); off += (size_t)DD * DD * 2;
  __bf16* wvb = (__bf16*)(ws + off); off += (size_t)DD * DD * 2;
  __bf16* qb = (__bf16*)(ws + off);  off += (size_t)BB * SS * DD * 2;  // 16 MB
  __bf16* kb = (__bf16*)(ws + off);  off += (size_t)BB * SS * DD * 2;  // 16 MB
  __bf16* vt = (__bf16*)(ws + off);  off += (size_t)BB * DD * SS * 2;  // 16 MB
  float* sb = (float*)(ws + off);    off += (size_t)BB * SS * SS * 4;  // 64 MB

  dim3 blk(256);

  k_cvt_bf16<<<2048, blk, 0, stream>>>(x, xb, BB * SS * DD);
  k_cvt_bf16<<<512, blk, 0, stream>>>(wq, wqb, DD * DD);
  k_cvt_bf16<<<512, blk, 0, stream>>>(wk, wkb, DD * DD);
  k_cvt_bf16<<<512, blk, 0, stream>>>(wv, wvb, DD * DD);

  dim3 gproj(BB * SS / 128, DD / 128, 1);
  k_qkv<<<gproj, blk, 0, stream>>>(xb, wqb, bq, qb, 0);
  k_qkv<<<gproj, blk, 0, stream>>>(xb, wkb, bk, kb, 1);
  k_qkv<<<gproj, blk, 0, stream>>>(xb, wvb, bv, vt, 2);

  k_scores<<<dim3(SS / 128, SS / 128, BB), blk, 0, stream>>>(qb, kb, sb);
  k_softmax<<<dim3(BB * SS), blk, 0, stream>>>(sb);
  k_pv<<<dim3(SS / 128, DD / 128, BB), blk, 0, stream>>>(sb, vt, out);
}
